// SymbolicCodecV5_59897613910106
// MI455X (gfx1250) — matrix operand order repaired and checked by reading, not compiled, not run on hardware
//
#include <hip/hip_runtime.h>

typedef __attribute__((ext_vector_type(16))) __bf16 v16bf;
typedef __attribute__((ext_vector_type(8)))  __bf16 v8bf;
typedef __attribute__((ext_vector_type(8)))  float  v8f;
typedef __attribute__((ext_vector_type(4)))  unsigned tdm_v4u;
typedef __attribute__((ext_vector_type(8)))  int      tdm_v8i;
typedef __attribute__((ext_vector_type(4)))  int      tdm_v4i;

#if defined(__has_builtin)
#  if __has_builtin(__builtin_amdgcn_tensor_load_to_lds)
#    define HAVE_TDM 1
#  endif
#endif
#ifndef HAVE_TDM
#  define HAVE_TDM 0
#endif

#define B_   8
#define P_   1024
#define D_   512
#define C_   8
#define K_   256
#define SUB_ 64
#define H_   512
#define PB_  16
#define T_   1023              // P-1
#define R_   (B_ * T_)         // 8184 prior rows
#define RP_  8192              // rows padded to multiple of 16
#define G3H_ 1536              // 3*H

#define CW_  0.25f
#define CBW_ 1.0f
#define UW_  0.1f
#define PW_  1.0f

// ---------------- CDNA5 helpers ----------------

// A-matrix 16x32 bf16 fragment: lane's 16 elements are K = 2*hi8+0..15 (contiguous,
// matching the B fragment order).
__device__ __forceinline__ v16bf load_a_frag(const __bf16* rowp, int hi8) {
  return *(const v16bf*)(rowp + 2 * hi8);
}
// B-matrix 32x16 fragment: lane's 16 elements are K = hi16+0..15 (contiguous).
__device__ __forceinline__ v16bf load_b_frag(const __bf16* colp, int hi16) {
  return *(const v16bf*)(colp + hi16);
}

__device__ __forceinline__ v8f wmma_bf16(v16bf a, v16bf b, v8f c) {
  return __builtin_amdgcn_wmma_f32_16x16x32_bf16(false, a, false, b, (short)0, c,
                                                 false, false);
}

// generic pointer to __shared__ carries the LDS offset in its low 32 bits
__device__ __forceinline__ unsigned lds_off(const void* p) {
  return (unsigned)(unsigned long long)p;
}
__device__ __forceinline__ void async_b128(unsigned lds, const void* g) {
  asm volatile("global_load_async_to_lds_b128 %0, %1, off" :: "v"(lds), "v"(g) : "memory");
}
__device__ __forceinline__ void async_wait0() {
  asm volatile("s_wait_asynccnt 0x0" ::: "memory");
}

// Tensor Data Mover: 2D tile (rows x cols bf16) global -> LDS, one issue per wave.
#if HAVE_TDM
__device__ __forceinline__ void tdm_load_2d(unsigned lds, const void* gptr,
                                            int cols, int rows, int stride_elts) {
  unsigned long long ga = (unsigned long long)gptr;
  tdm_v4u g0 = {0u, 0u, 0u, 0u};
  tdm_v8i g1 = {0, 0, 0, 0, 0, 0, 0, 0};
  tdm_v4i gz = {0, 0, 0, 0};
  tdm_v8i gz8 = {0, 0, 0, 0, 0, 0, 0, 0};
  g0[0] = 1u;                                            // count=1 (valid, user mode)
  g0[1] = lds;                                           // lds_addr
  g0[2] = (unsigned)ga;                                  // global_addr[31:0]
  g0[3] = (unsigned)((ga >> 32) & 0x01FFFFFFu) | (2u << 30);  // addr[56:32], type=2
  g1[0] = (int)(1u << 16);                               // data_size=1 (2 bytes)
  g1[1] = (int)(((unsigned)cols & 0xFFFFu) << 16);       // tensor_dim0[15:0]
  g1[2] = (int)((((unsigned)cols >> 16) & 0xFFFFu) |
                (((unsigned)rows & 0xFFFFu) << 16));     // dim0 hi | tensor_dim1 lo
  g1[3] = (int)(((unsigned)cols & 0xFFFFu) << 16);       // tile_dim0 = cols
  g1[4] = rows;                                          // tile_dim1
  g1[5] = stride_elts;                                   // tensor_dim0_stride
  __builtin_amdgcn_tensor_load_to_lds(g0, g1, gz, gz, gz8, 0);
  __builtin_amdgcn_s_wait_tensorcnt(0);
}
#endif

__device__ __forceinline__ float wred_sum(float v) {
  for (int m = 16; m; m >>= 1) v += __shfl_xor(v, m, 32);
  return v;
}

// sense-reversing inter-workgroup barrier (16 resident WGs)
__device__ __forceinline__ void gbar(int* cnt, volatile int* gen, int nwg) {
  __threadfence();
  __syncthreads();
  if (threadIdx.x == 0) {
    int g = *gen;
    __threadfence();
    if (atomicAdd(cnt, 1) == nwg - 1) {
      *cnt = 0;
      __threadfence();
      atomicAdd((int*)gen, 1);
    } else {
      while (*gen == g) { __builtin_amdgcn_s_sleep(1); }
    }
  }
  __syncthreads();
}

// ---------------- init / convert ----------------

__global__ void k_init(float* avg, float* accum) {
  int t = threadIdx.x;
  for (int i = t; i < C_ * K_; i += blockDim.x) avg[i] = 0.f;
  if (t < 64) accum[t] = 0.f;
}

__global__ void k_cvt_bf16(const float* __restrict__ src, __bf16* __restrict__ dst, int n) {
  for (int i = blockIdx.x * blockDim.x + threadIdx.x; i < n; i += gridDim.x * blockDim.x)
    dst[i] = (__bf16)src[i];
}

// ---------------- VQ: distances, argmin, softmax, soft, losses ----------------
// grid (512 row-tiles, C_), block 128 (4 waves)
__global__ void __launch_bounds__(128) k_vq(
    const float* __restrict__ latents, const float* __restrict__ patch_mask,
    const float* __restrict__ codebooks,
    float* __restrict__ out_ids, float* __restrict__ out_stq, float* __restrict__ out_probs,
    int* __restrict__ ids_i32, float* __restrict__ avg, float* __restrict__ accum)
{
  __shared__ __align__(16) float  sp[16][SUB_ + 4];   // f32 split tile (async dest)
  __shared__ __align__(32) __bf16 spb[16][SUB_];      // bf16 mirror (A frags)
  __shared__ __align__(32) __bf16 cbb[K_][SUB_];      // codebook [k][d] (dist B frags)
  __shared__ __align__(32) __bf16 cbt[SUB_][K_];      // codebook [d][k] (soft B frags)
  __shared__ __align__(32) __bf16 pbuf[16][K_];       // probs bf16 (soft A frags)
  __shared__ float d2[16][K_];                        // distances, reused for probs
  __shared__ float colsq[K_];
  __shared__ float rowsq[16];
  __shared__ float maskl[16];
  __shared__ int   idsl[16];

  const int tile = blockIdx.x, c = blockIdx.y;
  const int tid = threadIdx.x, w = tid >> 5, lane = tid & 31;
  const int m16 = lane & 15;
  const int hi8 = (lane >> 4) << 3;
  const int hi16 = (lane >> 4) << 4;

  // async-stage the 16x64 f32 split tile into LDS
  for (int e = tid; e < 16 * SUB_ / 4; e += 128) {
    int m = e >> 4, d4 = e & 15;
    int row = tile * 16 + m;
    async_b128(lds_off(&sp[m][d4 * 4]), latents + (size_t)row * D_ + c * SUB_ + d4 * 4);
  }
  // codebook[c]: f32 global -> dual-layout bf16 LDS
  const float* cbg = codebooks + (size_t)c * K_ * SUB_;
  for (int e = tid; e < (K_ * SUB_) / 4; e += 128) {
    int k = e >> 4, d4 = (e & 15) * 4;
    const float4 v = *(const float4*)(cbg + k * SUB_ + d4);
    __bf16 b0 = (__bf16)v.x, b1 = (__bf16)v.y, b2 = (__bf16)v.z, b3 = (__bf16)v.w;
    cbb[k][d4] = b0; cbb[k][d4 + 1] = b1; cbb[k][d4 + 2] = b2; cbb[k][d4 + 3] = b3;
    cbt[d4][k] = b0; cbt[d4 + 1][k] = b1; cbt[d4 + 2][k] = b2; cbt[d4 + 3][k] = b3;
  }
  async_wait0();
  __syncthreads();

  if (tid < 16) {
    int row = tile * 16 + tid;
    maskl[tid] = patch_mask[row];
    float s = 0.f;
    for (int d = 0; d < SUB_; ++d) { float v = sp[tid][d]; s += v * v; }
    rowsq[tid] = s;
  }
  for (int e = tid; e < 16 * SUB_; e += 128)
    spb[e >> 6][e & 63] = (__bf16)sp[e >> 6][e & 63];
  for (int k = tid; k < K_; k += 128) {
    float s = 0.f;
    for (int d = 0; d < SUB_; ++d) { float v = (float)cbb[k][d]; s += v * v; }
    colsq[k] = s;
  }
  __syncthreads();

  // d2 = ||s||^2 + ||c||^2 - 2 s.c   via WMMA, 4 N-tiles per wave
  for (int nt = w; nt < K_ / 16; nt += 4) {
    v8f acc = {};
#pragma unroll
    for (int ks = 0; ks < SUB_ / 32; ++ks) {
      v16bf af  = load_a_frag(&spb[m16][ks * 32], hi8);
      v16bf bfv = load_b_frag(&cbb[nt * 16 + m16][ks * 32], hi16);
      acc = wmma_bf16(af, bfv, acc);
    }
#pragma unroll
    for (int rr = 0; rr < 8; ++rr) {
      int m = rr + hi8, n = nt * 16 + m16;
      d2[m][n] = rowsq[m] + colsq[n] - 2.f * acc[rr];
    }
  }
  __syncthreads();

  // per-row argmin + softmax(-d2) (TEMP==1); wave handles 4 rows
  for (int rr = 0; rr < 4; ++rr) {
    int m = w * 4 + rr;
    float bv = 1e30f; int bi = K_;
    for (int k = lane; k < K_; k += 32) {
      float v = d2[m][k];
      if (v < bv || (v == bv && k < bi)) { bv = v; bi = k; }
    }
    for (int off = 16; off; off >>= 1) {
      float ov = __shfl_xor(bv, off, 32);
      int   oi = __shfl_xor(bi, off, 32);
      if (ov < bv || (ov == bv && oi < bi)) { bv = ov; bi = oi; }
    }
    float s = 0.f;
    for (int k = lane; k < K_; k += 32) s += __expf(-(d2[m][k] - bv));
    s = wred_sum(s);
    float inv = 1.f / s;
    int row = tile * 16 + m;
    float* pout = out_probs + (((size_t)row) * C_ + c) * K_;
    for (int k = lane; k < K_; k += 32) {
      float p = __expf(-(d2[m][k] - bv)) * inv;
      d2[m][k] = p;
      pbuf[m][k] = (__bf16)p;
      pout[k] = p;
    }
    if (lane == 0) {
      idsl[m] = bi;
      out_ids[(size_t)row * C_ + c] = (float)bi;
      ids_i32[(size_t)row * C_ + c] = bi;
    }
  }
  __syncthreads();

  // codebook-usage accumulation and st_quantized (= hard) output
  for (int k = tid; k < K_; k += 128) {
    float s = 0.f;
    for (int m = 0; m < 16; ++m) s += d2[m][k] * maskl[m];
    atomicAdd(&avg[c * K_ + k], s);
  }
  for (int e = tid; e < 16 * SUB_; e += 128) {
    int m = e >> 6, d = e & 63;
    int row = tile * 16 + m;
    out_stq[(size_t)row * D_ + c * SUB_ + d] = cbg[(size_t)idsl[m] * SUB_ + d];
  }

  // soft = probs @ cb  (one 16x16 N-tile per wave covers SUB=64), then MSE
  v8f sacc = {};
#pragma unroll
  for (int ks = 0; ks < K_ / 32; ++ks) {
    v16bf af  = load_a_frag(&pbuf[m16][ks * 32], hi8);
    v16bf bfv = load_b_frag(&cbt[w * 16 + m16][ks * 32], hi16);
    sacc = wmma_bf16(af, bfv, sacc);
  }
  float msel = 0.f;
#pragma unroll
  for (int rr = 0; rr < 8; ++rr) {
    int m = rr + hi8, n = w * 16 + m16;
    float diff = sp[m][n] - sacc[rr];
    msel += diff * diff;
  }
  msel = wred_sum(msel);
  if (lane == 0) atomicAdd(&accum[0], msel);
}

// ---------------- mask sums, usage loss, perplexity ----------------
__global__ void __launch_bounds__(256) k_stats(
    const float* __restrict__ patch_mask, const int* __restrict__ patch_ids,
    const float* __restrict__ avg, float* __restrict__ accum)
{
  __shared__ float red[16];
  __shared__ float s_active;
  __shared__ float s_usage[C_], s_ent[C_];
  int tid = threadIdx.x, w = tid >> 5, lane = tid & 31;

  float ms = 0.f;
  for (int i = tid; i < B_ * P_; i += 256) ms += patch_mask[i];
  ms = wred_sum(ms);
  if (lane == 0) red[w] = ms;
  __syncthreads();
  if (tid == 0) {
    float s = 0.f;
    for (int i = 0; i < 8; ++i) s += red[i];
    s_active = fmaxf(s, 1.f);
    accum[3] = s_active;
  }
  __syncthreads();

  float ps = 0.f, bs = 0.f;
  for (int i = tid; i < B_ * T_; i += 256) {
    int b = i / T_, t = i % T_;
    float pm = patch_mask[b * P_ + t] * patch_mask[b * P_ + t + 1];
    int cnt = 0;
    const int* pid = patch_ids + ((size_t)(b * P_ + t + 1)) * PB_;
    for (int j = 0; j < PB_; ++j) cnt += (pid[j] != 256);
    ps += pm; bs += (float)cnt * pm;
  }
  ps = wred_sum(ps); bs = wred_sum(bs);
  if (lane == 0) red[w] = ps;
  __syncthreads();
  if (tid == 0) { float p = 0.f; for (int i = 0; i < 8; ++i) p += red[i]; accum[4] = p; }
  __syncthreads();
  if (lane == 0) red[w] = bs;
  __syncthreads();
  if (tid == 0) { float p = 0.f; for (int i = 0; i < 8; ++i) p += red[i]; accum[5] = p; }

  // wave w handles group c = w
  float act = s_active;
  float us = 0.f, en = 0.f;
  for (int k = lane; k < K_; k += 32) {
    float a = avg[w * K_ + k] / act;
    us += a * (__logf(fmaxf(a, 1e-8f)) + __logf((float)K_));
    en += a * __logf(a + 1e-8f);
  }
  us = wred_sum(us); en = wred_sum(en);
  if (lane == 0) { s_usage[w] = us; s_ent[w] = en; }
  __syncthreads();
  if (tid == 0) {
    float u = 0.f, pp = 0.f;
    for (int cc = 0; cc < C_; ++cc) { u += s_usage[cc]; pp += __expf(-s_ent[cc]); }
    accum[1] = (u / C_) * UW_;
    accum[2] = pp / C_;
  }
}

// ---------------- proj: x = gather(emb) @ proj_w^T + b  ----------------
// grid (512, 4), block 256 (8 waves, 16 cols each)
__global__ void __launch_bounds__(256) k_proj(
    const __bf16* __restrict__ emb_bf, const __bf16* __restrict__ projw_bf,
    const float* __restrict__ proj_b, const int* __restrict__ ids_i32,
    __bf16* __restrict__ x_bf)
{
  __shared__ __align__(32) __bf16 A[16][256];   // 8 K-steps chunk
  __shared__ int rid[16][C_];                   // per-row symbol ids
  const int mt = blockIdx.x, nb = blockIdx.y;
  const int tid = threadIdx.x, w = tid >> 5, lane = tid & 31;
  const int m16 = lane & 15;
  const int hi8 = (lane >> 4) << 3;
  const int hi16 = (lane >> 4) << 4;
  const int n0 = nb * 128 + w * 16;

  if (tid < 16 * C_) {
    int m = tid >> 3, cg = tid & 7;
    int r = mt * 16 + m;
    int id = 0;
    if (r < R_) {
      int b = r / T_, t = r % T_;
      id = ids_i32[((size_t)(b * P_ + t)) * C_ + cg];
    }
    rid[m][cg] = id;
  }
  __syncthreads();

  v8f acc = {};
  for (int kc = 0; kc < C_ * H_; kc += 256) {
    for (int e = tid; e < 16 * 256; e += 256) {
      int m = e >> 8, j = e & 255;
      int jg = kc + j;
      int cg = jg >> 9, h = jg & 511;
      int r = mt * 16 + m;
      __bf16 v = (__bf16)0.f;
      if (r < R_) v = emb_bf[((size_t)(cg * K_ + rid[m][cg])) * H_ + h];
      A[m][j] = v;
    }
    __syncthreads();
#pragma unroll
    for (int ks = 0; ks < 8; ++ks) {
      const __bf16* bcol = projw_bf + (size_t)(n0 + m16) * (C_ * H_) + kc + ks * 32;
      v16bf af  = load_a_frag(&A[m16][ks * 32], hi8);
      v16bf bfv = load_b_frag(bcol, hi16);
      acc = wmma_bf16(af, bfv, acc);
    }
    __syncthreads();
  }
#pragma unroll
  for (int rr = 0; rr < 8; ++rr) {
    int m = rr + hi8, n = n0 + m16, r = mt * 16 + m;
    if (r < R_) x_bf[(size_t)r * H_ + n] = (__bf16)(acc[rr] + proj_b[n]);
  }
}

// ---------------- gx = in @ Wih^T + bih ----------------
// grid (512, 12), block 256; A tile staged via Tensor Data Mover (or async fallback)
__global__ void __launch_bounds__(256) k_gx(
    const __bf16* __restrict__ in_bf, const __bf16* __restrict__ w_bf,
    const float* __restrict__ bias, float* __restrict__ gx)
{
  __shared__ __align__(32) __bf16 A[16][H_];    // 16 KB
  const int mt = blockIdx.x, nb = blockIdx.y;
  const int tid = threadIdx.x, w = tid >> 5, lane = tid & 31;
  const int m16 = lane & 15;
  const int hi8 = (lane >> 4) << 3;
  const int hi16 = (lane >> 4) << 4;
  const int n0 = nb * 128 + w * 16;

#if HAVE_TDM
  if (tid < 32) {
    tdm_load_2d(lds_off(&A[0][0]), in_bf + (size_t)mt * 16 * H_, H_, 16, H_);
  }
#else
  for (int ch = tid; ch < 1024; ch += 256) {            // 1024 x 16B chunks
    int m = ch >> 6, o = ch & 63;
    int r = mt * 16 + m;
    async_b128(lds_off(&A[m][0]) + o * 16, in_bf + (size_t)r * H_ + o * 8);
  }
  async_wait0();
#endif
  __syncthreads();

  v8f acc = {};
#pragma unroll
  for (int ks = 0; ks < H_ / 32; ++ks) {
    const __bf16* bcol = w_bf + (size_t)(n0 + m16) * H_ + ks * 32;
    v16bf af  = load_a_frag(&A[m16][ks * 32], hi8);
    v16bf bfv = load_b_frag(bcol, hi16);
    acc = wmma_bf16(af, bfv, acc);
  }
#pragma unroll
  for (int rr = 0; rr < 8; ++rr) {
    int m = rr + hi8, n = n0 + m16, r = mt * 16 + m;
    gx[(size_t)r * G3H_ + n] = acc[rr] + bias[n];
  }
}

// ---------------- GRU recurrence (persistent, 16 WGs x 256) ----------------
// WG i owns h columns [32i,32i+32); per step computes 6 gh N-tiles via WMMA.
__global__ void __launch_bounds__(256) k_gru(
    const float* __restrict__ gx, const __bf16* __restrict__ whh_bf,
    const float* __restrict__ bhh, float* __restrict__ hglob,
    __bf16* __restrict__ hseq_bf, int* barcnt, int* bargen)
{
  __shared__ __align__(32) __bf16 hA[16][H_];   // 16 KB
  __shared__ float ght[16][96];
  const int wg = blockIdx.x;
  const int tid = threadIdx.x, w = tid >> 5, lane = tid & 31;
  const int m16 = lane & 15;
  const int hi8 = (lane >> 4) << 3;
  const int hi16 = (lane >> 4) << 4;

  // zero my columns of both h ping-pong buffers (all 16 padded rows)
  for (int e = tid; e < 2 * 16 * 32; e += 256) {
    int bb = e >> 9, m = (e >> 5) & 15, cc = e & 31;
    hglob[bb * (16 * H_) + m * H_ + wg * 32 + cc] = 0.f;
  }
  gbar(barcnt, bargen, 16);

  for (int t = 0; t < T_; ++t) {
    const float* hread = hglob + (t & 1) * (16 * H_);
    float* hwrite = hglob + ((t + 1) & 1) * (16 * H_);

    for (int e = tid; e < 16 * H_; e += 256)
      hA[e >> 9][e & 511] = (__bf16)hread[e];
    __syncthreads();

    if (w < 6) {
      int g = w >> 1;
      int ncol0 = g * H_ + wg * 32 + (w & 1) * 16;
      v8f acc = {};
#pragma unroll
      for (int ks = 0; ks < H_ / 32; ++ks) {
        const __bf16* bcol = whh_bf + (size_t)(ncol0 + m16) * H_ + ks * 32;
        v16bf af  = load_a_frag(&hA[m16][ks * 32], hi8);
        v16bf bfv = load_b_frag(bcol, hi16);
        acc = wmma_bf16(af, bfv, acc);
      }
#pragma unroll
      for (int rr = 0; rr < 8; ++rr) {
        int m = rr + hi8;
        ght[m][g * 32 + (w & 1) * 16 + m16] = acc[rr];
      }
    }
    __syncthreads();

    {
      int row = tid >> 5, lc = tid & 31;     // 8 batch rows x 32 cols
      int col = wg * 32 + lc;
      size_t gidx = ((size_t)row * T_ + t) * G3H_;
      float gr  = gx[gidx + col]            + ght[row][lc]      + bhh[col];
      float gz  = gx[gidx + H_ + col]       + ght[row][32 + lc] + bhh[H_ + col];
      float gn  = gx[gidx + 2 * H_ + col];
      float ghn = ght[row][64 + lc] + bhh[2 * H_ + col];
      float rg = 1.f / (1.f + __expf(-gr));
      float zg = 1.f / (1.f + __expf(-gz));
      float ng = tanhf(gn + rg * ghn);
      float hp = hread[row * H_ + col];
      float hn = (1.f - zg) * ng + zg * hp;
      hwrite[row * H_ + col] = hn;
      hseq_bf[((size_t)row * T_ + t) * H_ + col] = (__bf16)hn;
    }
    gbar(barcnt, bargen, 16);
  }
}

// ---------------- heads GEMM fused with log-softmax / KL / NLL / acc ----------------
// grid (512), block 256; two c-group passes of 1024 logit columns in LDS
__global__ void __launch_bounds__(256) k_heads(
    const __bf16* __restrict__ h_bf, const __bf16* __restrict__ headsw_bf,
    const float* __restrict__ heads_b, const float* __restrict__ patch_mask,
    const float* __restrict__ probs_out, const int* __restrict__ ids_i32,
    float* __restrict__ accum)
{
  __shared__ __align__(32) __bf16 A[16][H_];    // 16 KB
  __shared__ float lg[16][1024];                // 64 KB
  const int mt = blockIdx.x;
  const int tid = threadIdx.x, w = tid >> 5, lane = tid & 31;
  const int m16 = lane & 15;
  const int hi8 = (lane >> 4) << 3;
  const int hi16 = (lane >> 4) << 4;

  for (int ch = tid; ch < 1024; ch += 256) {
    int m = ch >> 6, o = ch & 63;
    int r = mt * 16 + m;
    async_b128(lds_off(&A[m][0]) + o * 16, h_bf + (size_t)r * H_ + o * 8);
  }
  async_wait0();
  __syncthreads();

  for (int pc = 0; pc < 2; ++pc) {
    for (int nt = w; nt < 64; nt += 8) {
      int ng0 = pc * 1024 + nt * 16;
      v8f acc = {};
#pragma unroll
      for (int ks = 0; ks < H_ / 32; ++ks) {
        const __bf16* bcol = headsw_bf + (size_t)(ng0 + m16) * H_ + ks * 32;
        v16bf af  = load_a_frag(&A[m16][ks * 32], hi8);
        v16bf bfv = load_b_frag(bcol, hi16);
        acc = wmma_bf16(af, bfv, acc);
      }
#pragma unroll
      for (int rr = 0; rr < 8; ++rr) {
        int m = rr + hi8;
        int nl = nt * 16 + m16;
        lg[m][nl] = acc[rr] + heads_b[pc * 1024 + nl];
      }
    }
    __syncthreads();

    // wave -> (c, row-half): 4 groups x 2 halves
    int c = pc * 4 + (w >> 1);
    int mbase = (w & 1) * 8;
    for (int mm = 0; mm < 8; ++mm) {
      int m = mbase + mm;
      int r = mt * 16 + m;
      if (r < R_) {
        int b = r / T_, t = r % T_;
        float pm = patch_mask[b * P_ + t] * patch_mask[b * P_ + t + 1];
        const float* lrow = &lg[m][(c & 3) * K_];
        float bv = -1e30f; int bi = 0;
        for (int k = lane; k < K_; k += 32) {
          float v = lrow[k];
          if (v > bv || (v == bv && k < bi)) { bv = v; bi = k; }
        }
        for (int off = 16; off; off >>= 1) {
          float ov = __shfl_xor(bv, off, 32);
          int   oi = __shfl_xor(bi, off, 32);
          if (ov > bv || (ov == bv && oi < bi)) { bv = ov; bi = oi; }
        }
        float se = 0.f;
        for (int k = lane; k < K_; k += 32) se += __expf(lrow[k] - bv);
        se = wred_sum(se);
        float lse = bv + __logf(se);
        const float* qrow = probs_out + (((size_t)(b * P_ + t + 1)) * C_ + c) * K_;
        float kl = 0.f;
        for (int k = lane; k < K_; k += 32) {
          float q = fmaxf(qrow[k], 1e-8f);
          kl += q * (__logf(q) - (lrow[k] - lse));
        }
        kl = wred_sum(kl);
        int tgt = ids_i32[((size_t)(b * P_ + t + 1)) * C_ + c];
        if (lane == 0) {
          float nll = -(lrow[tgt] - lse);
          atomicAdd(&accum[6 + c], kl * pm);
          atomicAdd(&accum[14], nll * pm);
          if (bi == tgt && pm > 0.f) atomicAdd(&accum[15], 1.f);
        }
      }
    }
    __syncthreads();
  }
}

// ---------------- scalar finalization ----------------
__global__ void k_final(const float* __restrict__ accum, float* __restrict__ outs) {
  if (threadIdx.x != 0 || blockIdx.x != 0) return;
  float mse_mean = accum[0] / (float)((size_t)B_ * P_ * D_);
  float pair = accum[4];
  float valid_w = fmaxf(pair, 1.f);
  float klsum = 0.f;
  for (int c = 0; c < C_; ++c) klsum += accum[6 + c] / valid_w;
  float group_tokens = fmaxf(pair * (float)C_, 1.f);
  outs[0] = mse_mean * CW_;                         // commitment_loss
  outs[1] = mse_mean * CBW_;                        // codebook_loss
  outs[2] = accum[1];                               // usage_loss
  outs[3] = accum[2];                               // perplexity
  outs[4] = (klsum / (float)C_) * PW_;              // prior_match_loss
  outs[5] = accum[14] / group_tokens;               // prior_ce_loss
  outs[6] = accum[15] / group_tokens;               // prior_accuracy
  outs[7] = accum[14] / 0.6931471805599453f / fmaxf(accum[5], 1.f);  // prior_bpb
}

// ---------------- host side ----------------

extern "C" void kernel_launch(void* const* d_in, const int* in_sizes, int n_in,
                              void* d_out, int out_size, void* d_ws, size_t ws_size,
                              hipStream_t stream) {
  (void)in_sizes; (void)n_in; (void)out_size; (void)ws_size;
  const float* latents    = (const float*)d_in[0];
  const float* patch_mask = (const float*)d_in[1];
  const int*   patch_ids  = (const int*)d_in[2];
  const float* codebooks  = (const float*)d_in[3];
  const float* group_emb  = (const float*)d_in[4];
  const float* proj_w     = (const float*)d_in[5];
  const float* proj_b     = (const float*)d_in[6];
  const float* gru_wih    = (const float*)d_in[7];
  const float* gru_whh    = (const float*)d_in[8];
  const float* gru_bih    = (const float*)d_in[9];
  const float* gru_bhh    = (const float*)d_in[10];
  const float* heads_w    = (const float*)d_in[11];
  const float* heads_b    = (const float*)d_in[12];

  float* out_ids   = (float*)d_out;
  float* out_stq   = out_ids + (size_t)B_ * P_ * C_;
  float* out_probs = out_stq + (size_t)B_ * P_ * D_;
  float* out_scal  = out_probs + (size_t)B_ * P_ * C_ * K_;

  char* wp = (char*)d_ws;
  auto alloc = [&](size_t bytes) {
    char* p = wp;
    wp += (bytes + 255) & ~(size_t)255;
    return (void*)p;
  };
  __bf16* emb_bf   = (__bf16*)alloc((size_t)C_ * K_ * H_ * 2);
  __bf16* projw_bf = (__bf16*)alloc((size_t)H_ * C_ * H_ * 2);
  __bf16* wih_bf   = (__bf16*)alloc((size_t)2 * G3H_ * H_ * 2);
  __bf16* whh_bf   = (__bf16*)alloc((size_t)2 * G3H_ * H_ * 2);
  __bf16* hw_bf    = (__bf16*)alloc((size_t)C_ * K_ * H_ * 2);
  __bf16* x_bf     = (__bf16*)alloc((size_t)RP_ * H_ * 2);
  __bf16* h0_bf    = (__bf16*)alloc((size_t)RP_ * H_ * 2);
  __bf16* h1_bf    = (__bf16*)alloc((size_t)RP_ * H_ * 2);
  float*  gx       = (float*)alloc((size_t)RP_ * G3H_ * 4);
  float*  hglob    = (float*)alloc((size_t)2 * 16 * H_ * 4);
  float*  avg      = (float*)alloc((size_t)C_ * K_ * 4);
  float*  accum    = (float*)alloc(64 * 4);
  int*    ids_i32  = (int*)alloc((size_t)B_ * P_ * C_ * 4);

  k_init<<<1, 256, 0, stream>>>(avg, accum);
  k_cvt_bf16<<<512, 256, 0, stream>>>(group_emb, emb_bf, C_ * K_ * H_);
  k_cvt_bf16<<<512, 256, 0, stream>>>(proj_w, projw_bf, H_ * C_ * H_);
  k_cvt_bf16<<<512, 256, 0, stream>>>(gru_wih, wih_bf, 2 * G3H_ * H_);
  k_cvt_bf16<<<512, 256, 0, stream>>>(gru_whh, whh_bf, 2 * G3H_ * H_);
  k_cvt_bf16<<<512, 256, 0, stream>>>(heads_w, hw_bf, C_ * K_ * H_);

  k_vq<<<dim3(512, C_), 128, 0, stream>>>(latents, patch_mask, codebooks,
                                          out_ids, out_stq, out_probs,
                                          ids_i32, avg, accum);
  k_stats<<<1, 256, 0, stream>>>(patch_mask, patch_ids, avg, accum);

  k_proj<<<dim3(512, 4), 256, 0, stream>>>(emb_bf, projw_bf, proj_b, ids_i32, x_bf);

  // layer 0
  k_gx<<<dim3(512, 12), 256, 0, stream>>>(x_bf, wih_bf, gru_bih, gx);
  k_gru<<<16, 256, 0, stream>>>(gx, whh_bf, gru_bhh, hglob, h0_bf,
                                (int*)(accum + 16), (int*)(accum + 17));
  // layer 1
  k_gx<<<dim3(512, 12), 256, 0, stream>>>(h0_bf, wih_bf + (size_t)G3H_ * H_,
                                          gru_bih + G3H_, gx);
  k_gru<<<16, 256, 0, stream>>>(gx, whh_bf + (size_t)G3H_ * H_, gru_bhh + G3H_,
                                hglob, h1_bf,
                                (int*)(accum + 16), (int*)(accum + 17));

  k_heads<<<512, 256, 0, stream>>>(h1_bf, hw_bf, heads_b, patch_mask,
                                   out_probs, ids_i32, accum);
  k_final<<<1, 32, 0, stream>>>(accum, out_scal);
}
